// AttentionSR_84722524881144
// MI455X (gfx1250) — compile-verified, matrix operand order checked by reading, not run on hardware
//
#include <hip/hip_runtime.h>
#include <hip/hip_bf16.h>

typedef __attribute__((ext_vector_type(16))) _Float16 v16h;
typedef __attribute__((ext_vector_type(8)))  _Float16 v8h;
typedef __attribute__((ext_vector_type(8)))  float    v8f;

// ---------------- constants ----------------
#define CDIM    512
#define KSR     2048      // 4*512 patch GEMM K

// workspace layout (bytes)
#define OFF_XH    ((size_t)0)                    // 16384*512 f16 = 16 MiB
#define OFF_WQH   (OFF_XH   + (size_t)16777216)  // 512*512  f16
#define OFF_WKVH  (OFF_WQH  + (size_t)524288)    // 1024*512 f16
#define OFF_WPH   (OFF_WKVH + (size_t)1048576)   // 512*512  f16
#define OFF_WSRH  (OFF_WPH  + (size_t)524288)    // 512*2048 f16
#define OFF_XSR   (OFF_WSRH + (size_t)2097152)   // 4096*512 f32
#define OFF_XLN   (OFF_XSR  + (size_t)8388608)   // 4096*512 f16
#define OFF_QH    (OFF_XLN  + (size_t)4194304)   // 32*4096*64 f16
#define OFF_KH    (OFF_QH   + (size_t)16777216)  // 32*1024*64 f16
#define OFF_VT    (OFF_KH   + (size_t)4194304)   // 32*64*1024 f16
#define OFF_AOH   (OFF_VT   + (size_t)4194304)   // 16384*512 f16

// ---------------- helpers ----------------
// Load a 16x16x32 WMMA A/B fragment from a row-major matrix.
// Element e of lane L holds K = k0 + (e/8)*16 + (L/16)*8 + (e%8); M/N = L%16.
__device__ __forceinline__ v16h load_frag(const _Float16* __restrict__ row,
                                          int k0, int lh /* (lane/16)*8 */) {
  v8h lo = *(const v8h*)(row + k0 + lh);
  v8h hi = *(const v8h*)(row + k0 + lh + 16);
  return __builtin_shufflevector(lo, hi, 0,1,2,3,4,5,6,7,8,9,10,11,12,13,14,15);
}

__device__ __forceinline__ v8f wmma_f16(v16h a, v16h b, v8f c) {
  return __builtin_amdgcn_wmma_f32_16x16x32_f16(false, a, false, b, (short)0, c,
                                                false, false);
}

// Software-pipelined 32x64 GEMM core over K=512 (row-major A rows, W rows).
__device__ __forceinline__ void gemm_core_512(const _Float16* __restrict__ ar0,
                                              const _Float16* __restrict__ ar1,
                                              const _Float16* __restrict__ br0,
                                              const _Float16* __restrict__ br1,
                                              const _Float16* __restrict__ br2,
                                              const _Float16* __restrict__ br3,
                                              int lh, v8f (&c)[2][4]) {
  v16h a0 = load_frag(ar0, 0, lh), a1 = load_frag(ar1, 0, lh);
  v16h b0 = load_frag(br0, 0, lh), b1 = load_frag(br1, 0, lh);
  v16h b2 = load_frag(br2, 0, lh), b3 = load_frag(br3, 0, lh);
  for (int k0 = 32; k0 < 512; k0 += 32) {
    // prefetch next K-step while current WMMAs execute
    v16h a0n = load_frag(ar0, k0, lh), a1n = load_frag(ar1, k0, lh);
    v16h b0n = load_frag(br0, k0, lh), b1n = load_frag(br1, k0, lh);
    v16h b2n = load_frag(br2, k0, lh), b3n = load_frag(br3, k0, lh);
    c[0][0] = wmma_f16(a0, b0, c[0][0]);  c[1][0] = wmma_f16(a1, b0, c[1][0]);
    c[0][1] = wmma_f16(a0, b1, c[0][1]);  c[1][1] = wmma_f16(a1, b1, c[1][1]);
    c[0][2] = wmma_f16(a0, b2, c[0][2]);  c[1][2] = wmma_f16(a1, b2, c[1][2]);
    c[0][3] = wmma_f16(a0, b3, c[0][3]);  c[1][3] = wmma_f16(a1, b3, c[1][3]);
    a0 = a0n; a1 = a1n; b0 = b0n; b1 = b1n; b2 = b2n; b3 = b3n;
  }
  c[0][0] = wmma_f16(a0, b0, c[0][0]);  c[1][0] = wmma_f16(a1, b0, c[1][0]);
  c[0][1] = wmma_f16(a0, b1, c[0][1]);  c[1][1] = wmma_f16(a1, b1, c[1][1]);
  c[0][2] = wmma_f16(a0, b2, c[0][2]);  c[1][2] = wmma_f16(a1, b2, c[1][2]);
  c[0][3] = wmma_f16(a0, b3, c[0][3]);  c[1][3] = wmma_f16(a1, b3, c[1][3]);
}

// ---------------- conversion / packing ----------------
__global__ void cvt_f32_f16(const float* __restrict__ s, _Float16* __restrict__ d, int n) {
  int i = blockIdx.x * 256 + threadIdx.x;
  if (i < n) d[i] = (_Float16)s[i];
}

// Wsr [c_out][c_in][2][2] -> Wsr_h [c_out][k=2048], k = (kh*2+kw)*512 + c_in
__global__ void pack_wsr(const float* __restrict__ w, _Float16* __restrict__ d) {
  int i = blockIdx.x * 256 + threadIdx.x;
  if (i < 512 * 2048) {
    int co = i >> 11, k = i & 2047;
    int s = k >> 9, ci = k & 511;
    d[i] = (_Float16)w[(co * 512 + ci) * 4 + s];
  }
}

// ---------------- GEMM: Q projection (32x64 tiles) ----------------
__global__ void __launch_bounds__(32) gemm_q(const _Float16* __restrict__ A,
                                             const _Float16* __restrict__ W,
                                             _Float16* __restrict__ Q) {
  int mt = blockIdx.x, nt = blockIdx.y;
  int lane = threadIdx.x, lr = lane & 15, lh = (lane >> 4) * 8;
  const _Float16* ar0 = A + (size_t)(mt * 32 + lr) * CDIM;
  const _Float16* ar1 = A + (size_t)(mt * 32 + 16 + lr) * CDIM;
  const _Float16* br0 = W + (size_t)(nt * 64 + 0  + lr) * CDIM;
  const _Float16* br1 = W + (size_t)(nt * 64 + 16 + lr) * CDIM;
  const _Float16* br2 = W + (size_t)(nt * 64 + 32 + lr) * CDIM;
  const _Float16* br3 = W + (size_t)(nt * 64 + 48 + lr) * CDIM;
  v8f c[2][4];
  for (int s = 0; s < 2; s++) for (int t = 0; t < 4; t++)
    for (int r = 0; r < 8; r++) c[s][t][r] = 0.0f;
  gemm_core_512(ar0, ar1, br0, br1, br2, br3, lh, c);
  for (int s = 0; s < 2; s++)
    for (int t = 0; t < 4; t++)
      for (int r = 0; r < 8; r++) {
        int m = mt * 32 + s * 16 + r + 8 * (lane >> 4);
        int b = m >> 12, nq = m & 4095;
        int col = nt * 64 + t * 16 + lr;
        int h = col >> 6, dd = col & 63;
        Q[(size_t)(((b << 3) + h) * 4096 + nq) * 64 + dd] = (_Float16)c[s][t][r];
      }
}

// ---------------- GEMM: SR patch-merge conv (stride2,k2) + bias ----------------
__global__ void __launch_bounds__(32) gemm_sr(const _Float16* __restrict__ xh,
                                              const _Float16* __restrict__ W,
                                              const float* __restrict__ bsr,
                                              float* __restrict__ out) {
  int mt = blockIdx.x, nt = blockIdx.y;
  int lane = threadIdx.x, lr = lane & 15, lh = (lane >> 4) * 8;
  int m0 = mt * 32 + lr, m1 = m0 + 16;
  int b0_ = m0 >> 10, p0 = m0 & 1023, oh0 = p0 >> 5, ow0 = p0 & 31;
  int b1_ = m1 >> 10, p1 = m1 & 1023, oh1 = p1 >> 5, ow1 = p1 & 31;
  const _Float16* br0 = W + (size_t)(nt * 64 + 0  + lr) * KSR;
  const _Float16* br1 = W + (size_t)(nt * 64 + 16 + lr) * KSR;
  const _Float16* br2 = W + (size_t)(nt * 64 + 32 + lr) * KSR;
  const _Float16* br3 = W + (size_t)(nt * 64 + 48 + lr) * KSR;
  v8f c[2][4];
  for (int s = 0; s < 2; s++) for (int t = 0; t < 4; t++)
    for (int r = 0; r < 8; r++) c[s][t][r] = 0.0f;

  // patch A-row addressing: k -> (kh,kw) selects source token row
  #define SR_AROW(bb, oh, ow, k0) \
    (xh + (size_t)((bb) * 4096 + (2*(oh) + (((k0) >> 10) & 1)) * 64 \
                   + 2*(ow) + (((k0) >> 9) & 1)) * CDIM)

  v16h a0 = load_frag(SR_AROW(b0_, oh0, ow0, 0), 0, lh);
  v16h a1 = load_frag(SR_AROW(b1_, oh1, ow1, 0), 0, lh);
  v16h b0 = load_frag(br0, 0, lh), b1 = load_frag(br1, 0, lh);
  v16h b2 = load_frag(br2, 0, lh), b3 = load_frag(br3, 0, lh);
  for (int k0 = 32; k0 < KSR; k0 += 32) {
    v16h a0n = load_frag(SR_AROW(b0_, oh0, ow0, k0), k0 & 511, lh);
    v16h a1n = load_frag(SR_AROW(b1_, oh1, ow1, k0), k0 & 511, lh);
    v16h b0n = load_frag(br0, k0, lh), b1n = load_frag(br1, k0, lh);
    v16h b2n = load_frag(br2, k0, lh), b3n = load_frag(br3, k0, lh);
    c[0][0] = wmma_f16(a0, b0, c[0][0]);  c[1][0] = wmma_f16(a1, b0, c[1][0]);
    c[0][1] = wmma_f16(a0, b1, c[0][1]);  c[1][1] = wmma_f16(a1, b1, c[1][1]);
    c[0][2] = wmma_f16(a0, b2, c[0][2]);  c[1][2] = wmma_f16(a1, b2, c[1][2]);
    c[0][3] = wmma_f16(a0, b3, c[0][3]);  c[1][3] = wmma_f16(a1, b3, c[1][3]);
    a0 = a0n; a1 = a1n; b0 = b0n; b1 = b1n; b2 = b2n; b3 = b3n;
  }
  c[0][0] = wmma_f16(a0, b0, c[0][0]);  c[1][0] = wmma_f16(a1, b0, c[1][0]);
  c[0][1] = wmma_f16(a0, b1, c[0][1]);  c[1][1] = wmma_f16(a1, b1, c[1][1]);
  c[0][2] = wmma_f16(a0, b2, c[0][2]);  c[1][2] = wmma_f16(a1, b2, c[1][2]);
  c[0][3] = wmma_f16(a0, b3, c[0][3]);  c[1][3] = wmma_f16(a1, b3, c[1][3]);
  #undef SR_AROW

  for (int s = 0; s < 2; s++)
    for (int t = 0; t < 4; t++)
      for (int r = 0; r < 8; r++) {
        int mm = mt * 32 + s * 16 + r + 8 * (lane >> 4);
        int col = nt * 64 + t * 16 + lr;
        out[(size_t)mm * CDIM + col] = c[s][t][r] + bsr[col];
      }
}

// ---------------- LayerNorm over 512-wide rows ----------------
__global__ void __launch_bounds__(256) ln_rows(const float* __restrict__ x,
                                               const float* __restrict__ g,
                                               const float* __restrict__ bt,
                                               _Float16* __restrict__ out) {
  __shared__ float red[256];
  int row = blockIdx.x, t = threadIdx.x;
  float v0 = x[(size_t)row * 512 + t];
  float v1 = x[(size_t)row * 512 + 256 + t];
  red[t] = v0 + v1;
  __syncthreads();
  for (int s = 128; s > 0; s >>= 1) { if (t < s) red[t] += red[t + s]; __syncthreads(); }
  float mu = red[0] * (1.0f / 512.0f);
  __syncthreads();
  float d0 = v0 - mu, d1 = v1 - mu;
  red[t] = d0 * d0 + d1 * d1;
  __syncthreads();
  for (int s = 128; s > 0; s >>= 1) { if (t < s) red[t] += red[t + s]; __syncthreads(); }
  float rstd = rsqrtf(red[0] * (1.0f / 512.0f) + 1e-5f);
  out[(size_t)row * 512 + t]       = (_Float16)(d0 * rstd * g[t] + bt[t]);
  out[(size_t)row * 512 + 256 + t] = (_Float16)(d1 * rstd * g[256 + t] + bt[256 + t]);
}

// ---------------- GEMM: KV projection ----------------
__global__ void __launch_bounds__(32) gemm_kv(const _Float16* __restrict__ A,
                                              const _Float16* __restrict__ W,
                                              _Float16* __restrict__ Kh,
                                              _Float16* __restrict__ Vt) {
  int mt = blockIdx.x, nt = blockIdx.y;
  int lane = threadIdx.x, lr = lane & 15, lh = (lane >> 4) * 8;
  const _Float16* ar0 = A + (size_t)(mt * 32 + lr) * CDIM;
  const _Float16* ar1 = A + (size_t)(mt * 32 + 16 + lr) * CDIM;
  const _Float16* br0 = W + (size_t)(nt * 64 + 0  + lr) * CDIM;
  const _Float16* br1 = W + (size_t)(nt * 64 + 16 + lr) * CDIM;
  const _Float16* br2 = W + (size_t)(nt * 64 + 32 + lr) * CDIM;
  const _Float16* br3 = W + (size_t)(nt * 64 + 48 + lr) * CDIM;
  v8f c[2][4];
  for (int s = 0; s < 2; s++) for (int t = 0; t < 4; t++)
    for (int r = 0; r < 8; r++) c[s][t][r] = 0.0f;
  gemm_core_512(ar0, ar1, br0, br1, br2, br3, lh, c);
  for (int s = 0; s < 2; s++)
    for (int t = 0; t < 4; t++)
      for (int r = 0; r < 8; r++) {
        int m = mt * 32 + s * 16 + r + 8 * (lane >> 4);  // 0..4095 over B*N'
        int b = m >> 10, mp = m & 1023;
        int col = nt * 64 + t * 16 + lr;                 // 0..1023
        _Float16 v = (_Float16)c[s][t][r];
        if (col < 512) {
          int h = col >> 6, dd = col & 63;
          Kh[(size_t)(((b << 3) + h) * 1024 + mp) * 64 + dd] = v;
        } else {
          int cc = col - 512;
          int h = cc >> 6, dd = cc & 63;
          Vt[(size_t)(((b << 3) + h) * 64 + dd) * 1024 + mp] = v;
        }
      }
}

// ---------------- Flash attention: one wave = 16 queries of one (b,h) ----------------
struct KF { v16h f[4]; };
__device__ __forceinline__ KF load_kf(const _Float16* __restrict__ Kbh, int n0,
                                      int lr, int lh) {
  KF k;
  const _Float16* kr0 = Kbh + (size_t)(n0 + lr) * 64;
  const _Float16* kr1 = Kbh + (size_t)(n0 + 16 + lr) * 64;
  k.f[0] = load_frag(kr0, 0, lh);  k.f[1] = load_frag(kr0, 32, lh);
  k.f[2] = load_frag(kr1, 0, lh);  k.f[3] = load_frag(kr1, 32, lh);
  return k;
}

__global__ void __launch_bounds__(32) attn(const _Float16* __restrict__ Qh,
                                           const _Float16* __restrict__ Kh,
                                           const _Float16* __restrict__ Vt,
                                           _Float16* __restrict__ AO) {
  int mt = blockIdx.x;          // query tile 0..255
  int bh = blockIdx.y;          // 0..31
  int b = bh >> 3, h = bh & 7;
  int lane = threadIdx.x, lr = lane & 15, lhh = lane >> 4, lh = lhh * 8;

  __shared__ alignas(16) float    S[16][32];
  __shared__ alignas(16) _Float16 P[16][32];
  __shared__ alignas(16) float    fs[16];
  __shared__ alignas(16) float    ls[16];

  const _Float16* qrow = Qh + (size_t)(bh * 4096 + mt * 16 + lr) * 64;
  v16h a0 = load_frag(qrow, 0, lh);
  v16h a1 = load_frag(qrow, 32, lh);
  const _Float16* Kbh = Kh + (size_t)bh * 1024 * 64;
  const _Float16* Vbh = Vt + (size_t)bh * 64 * 1024;

  v8f o[4];
  for (int t = 0; t < 4; t++) for (int r = 0; r < 8; r++) o[t][r] = 0.0f;
  float mrun = -1e30f, lrun = 0.0f;        // live in lanes 0..15 (row = lr)
  const float scale = 0.125f;              // 1/sqrt(64)

  KF kc = load_kf(Kbh, 0, lr, lh);         // prologue: K fragments for chunk 0
  for (int j = 0; j < 32; ++j) {
    int n0 = j * 32;
    // ---- V fragments for this chunk (independent of softmax; issue early) ----
    v16h vb[4];
    for (int t = 0; t < 4; t++)
      vb[t] = load_frag(Vbh + (size_t)(t * 16 + lr) * 1024 + n0, 0, lh);
    // ---- scores: S(16x32) = Q(16x64) @ K^T ----
    v8f s0, s1;
    for (int r = 0; r < 8; r++) { s0[r] = 0.0f; s1[r] = 0.0f; }
    s0 = wmma_f16(a0, kc.f[0], s0);
    s0 = wmma_f16(a1, kc.f[1], s0);
    s1 = wmma_f16(a0, kc.f[2], s1);
    s1 = wmma_f16(a1, kc.f[3], s1);
    // ---- prefetch next chunk's K fragments (overlaps softmax latency) ----
    int n0n = (j + 1 < 32) ? n0 + 32 : n0;
    KF kn = load_kf(Kbh, n0n, lr, lh);
    // ---- scores to LDS ----
    for (int r = 0; r < 8; r++) {
      S[r + 8 * lhh][lr]      = s0[r];
      S[r + 8 * lhh][16 + lr] = s1[r];
    }
    __syncthreads();
    // ---- per-row online softmax (lanes 0..15 own one row each) ----
    if (lane < 16) {
      float mx = mrun;
      float sv[32];
      #pragma unroll
      for (int i = 0; i < 32; i++) { sv[i] = S[lr][i] * scale; mx = fmaxf(mx, sv[i]); }
      float f = __expf(mrun - mx);
      float sum = 0.0f;
      #pragma unroll
      for (int i = 0; i < 32; i++) {
        float p = __expf(sv[i] - mx);
        sum += p;
        P[lr][i] = (_Float16)p;
      }
      lrun = lrun * f + sum;
      mrun = mx;
      fs[lr] = f;
    }
    __syncthreads();
    // ---- rescale accumulator ----
    for (int r = 0; r < 8; r++) {
      float f = fs[r + 8 * lhh];
      for (int t = 0; t < 4; t++) o[t][r] *= f;
    }
    // ---- O += P(16x32) @ V(32x64) ----
    v16h pa;
    {
      v8h lo = *(const v8h*)&P[lr][lh];
      v8h hi = *(const v8h*)&P[lr][16 + lh];
      pa = __builtin_shufflevector(lo, hi, 0,1,2,3,4,5,6,7,8,9,10,11,12,13,14,15);
    }
    for (int t = 0; t < 4; t++) o[t] = wmma_f16(pa, vb[t], o[t]);
    __syncthreads();   // protect S/P/fs before next chunk
    kc = kn;
  }
  if (lane < 16) ls[lr] = lrun;
  __syncthreads();
  for (int r = 0; r < 8; r++) {
    float inv = 1.0f / ls[r + 8 * lhh];
    int nq = mt * 16 + r + 8 * lhh;
    for (int t = 0; t < 4; t++) {
      int col = h * 64 + t * 16 + lr;
      AO[(size_t)(b * 4096 + nq) * CDIM + col] = (_Float16)(o[t][r] * inv);
    }
  }
}

// ---------------- GEMM: output projection + bias (f32 out) ----------------
__global__ void __launch_bounds__(32) gemm_out(const _Float16* __restrict__ A,
                                               const _Float16* __restrict__ W,
                                               const float* __restrict__ bp,
                                               float* __restrict__ out) {
  int mt = blockIdx.x, nt = blockIdx.y;
  int lane = threadIdx.x, lr = lane & 15, lh = (lane >> 4) * 8;
  const _Float16* ar0 = A + (size_t)(mt * 32 + lr) * CDIM;
  const _Float16* ar1 = A + (size_t)(mt * 32 + 16 + lr) * CDIM;
  const _Float16* br0 = W + (size_t)(nt * 64 + 0  + lr) * CDIM;
  const _Float16* br1 = W + (size_t)(nt * 64 + 16 + lr) * CDIM;
  const _Float16* br2 = W + (size_t)(nt * 64 + 32 + lr) * CDIM;
  const _Float16* br3 = W + (size_t)(nt * 64 + 48 + lr) * CDIM;
  v8f c[2][4];
  for (int s = 0; s < 2; s++) for (int t = 0; t < 4; t++)
    for (int r = 0; r < 8; r++) c[s][t][r] = 0.0f;
  gemm_core_512(ar0, ar1, br0, br1, br2, br3, lh, c);
  for (int s = 0; s < 2; s++)
    for (int t = 0; t < 4; t++)
      for (int r = 0; r < 8; r++) {
        int m = mt * 32 + s * 16 + r + 8 * (lane >> 4);
        int col = nt * 64 + t * 16 + lr;
        out[(size_t)m * CDIM + col] = c[s][t][r] + bp[col];
      }
}

// ---------------- host ----------------
extern "C" void kernel_launch(void* const* d_in, const int* in_sizes, int n_in,
                              void* d_out, int out_size, void* d_ws, size_t ws_size,
                              hipStream_t stream) {
  const float* x    = (const float*)d_in[0];
  // d_in[1], d_in[2] are H, W (compile-time 64 here)
  const float* Wq   = (const float*)d_in[3];
  const float* Wkv  = (const float*)d_in[4];
  const float* Wsr  = (const float*)d_in[5];
  const float* bsr  = (const float*)d_in[6];
  const float* ln_g = (const float*)d_in[7];
  const float* ln_b = (const float*)d_in[8];
  const float* Wp   = (const float*)d_in[9];
  const float* bp   = (const float*)d_in[10];
  float* out = (float*)d_out;

  char* ws = (char*)d_ws;
  _Float16* xh   = (_Float16*)(ws + OFF_XH);
  _Float16* wqh  = (_Float16*)(ws + OFF_WQH);
  _Float16* wkvh = (_Float16*)(ws + OFF_WKVH);
  _Float16* wph  = (_Float16*)(ws + OFF_WPH);
  _Float16* wsrh = (_Float16*)(ws + OFF_WSRH);
  float*    xsr  = (float*)   (ws + OFF_XSR);
  _Float16* xln  = (_Float16*)(ws + OFF_XLN);
  _Float16* qh   = (_Float16*)(ws + OFF_QH);
  _Float16* kh   = (_Float16*)(ws + OFF_KH);
  _Float16* vt   = (_Float16*)(ws + OFF_VT);
  _Float16* aoh  = (_Float16*)(ws + OFF_AOH);

  // 1) convert / pack
  cvt_f32_f16<<<(16384 * 512 + 255) / 256, 256, 0, stream>>>(x, xh, 16384 * 512);
  cvt_f32_f16<<<(512 * 512 + 255) / 256, 256, 0, stream>>>(Wq, wqh, 512 * 512);
  cvt_f32_f16<<<(1024 * 512 + 255) / 256, 256, 0, stream>>>(Wkv, wkvh, 1024 * 512);
  cvt_f32_f16<<<(512 * 512 + 255) / 256, 256, 0, stream>>>(Wp, wph, 512 * 512);
  pack_wsr<<<(512 * 2048 + 255) / 256, 256, 0, stream>>>(Wsr, wsrh);

  // 2) Q projection: 16384x512x512
  gemm_q<<<dim3(16384 / 32, 512 / 64), 32, 0, stream>>>(xh, wqh, qh);

  // 3) SR patch-merge GEMM: 4096x512x2048 (+bias)
  gemm_sr<<<dim3(4096 / 32, 512 / 64), 32, 0, stream>>>(xh, wsrh, bsr, xsr);

  // 4) LayerNorm rows -> f16
  ln_rows<<<4096, 256, 0, stream>>>(xsr, ln_g, ln_b, xln);

  // 5) KV projection: 4096x1024x512
  gemm_kv<<<dim3(4096 / 32, 1024 / 64), 32, 0, stream>>>(xln, wkvh, kh, vt);

  // 6) flash attention per (b,h): 32 heads x 256 query-tiles
  attn<<<dim3(256, 32), 32, 0, stream>>>(qh, kh, vt, aoh);

  // 7) output projection: 16384x512x512 (+bias, f32)
  gemm_out<<<dim3(16384 / 32, 512 / 64), 32, 0, stream>>>(aoh, wph, bp, out);
}
